// Temporal_Attention_13563506721058
// MI455X (gfx1250) — matrix operand order repaired and checked by reading, not compiled, not run on hardware
//
#include <hip/hip_runtime.h>
#include <hip/hip_bf16.h>

typedef __attribute__((ext_vector_type(16))) __bf16 v16bf;
typedef __attribute__((ext_vector_type(8)))  __bf16 v8bf;
typedef __attribute__((ext_vector_type(8)))  float  v8f;
typedef __attribute__((ext_vector_type(4)))  unsigned int v4u;
typedef __attribute__((ext_vector_type(8)))  int v8i;
typedef __attribute__((ext_vector_type(4)))  int v4i;

#define HEADS 6
#define NB    64
#define CCH   192
#define TT    64
#define VV    25
#define DH    32
#define BB    (NB*VV)          // 1600
#define O3    (3*CCH)          // 576

// workspace layout (in __bf16 elements)
// q,k slots: [b][slot][t][dd]; v slots (12..17): [b][slot][dd][t]
#define OFF_QKV   ((size_t)0)                               // [B][18][64*32]
#define OFF_ATTN  (OFF_QKV  + (size_t)BB*18*TT*DH)          // [B][64][192]
#define OFF_WQKV  (OFF_ATTN + (size_t)BB*TT*CCH)            // [576][192]
#define OFF_WOUT  (OFF_WQKV + (size_t)O3*CCH)               // [192][192]
#define OFF_REL   (OFF_WOUT + (size_t)CCH*CCH)              // [128][32]

static __device__ __forceinline__ v8f wmma_bf16(v16bf a, v16bf b, v8f c) {
  return __builtin_amdgcn_wmma_f32_16x16x32_bf16(false, a, false, b, (short)0, c,
                                                 false, false);
}
static __device__ __forceinline__ v16bf cat8(v8bf lo, v8bf hi) {
  return __builtin_shufflevector(lo, hi, 0, 1, 2, 3, 4, 5, 6, 7,
                                 8, 9, 10, 11, 12, 13, 14, 15);
}

// ---------------- prep: f32 -> bf16 for weights + rel table ----------------
__global__ __launch_bounds__(256) void prep_kernel(
    const float* __restrict__ wqkv, const float* __restrict__ wout,
    const float* __restrict__ rel, __bf16* __restrict__ ws) {
  int i = blockIdx.x * 256 + threadIdx.x;
  if (i < O3 * CCH)   ws[OFF_WQKV + i] = (__bf16)wqkv[i];
  if (i < CCH * CCH)  ws[OFF_WOUT + i] = (__bf16)wout[i];
  if (i < 128 * DH)   ws[OFF_REL  + i] = (__bf16)rel[i];   // rel_emb rows 0..127
}

// ---------------- QKV projection: 576x192 x 192x64 per b -------------------
__global__ __launch_bounds__(256) void qkv_gemm(
    const float* __restrict__ x, const float* __restrict__ bias,
    __bf16* __restrict__ ws) {
  const int b = blockIdx.x;
  const int n = b / VV, vv = b % VV;
  const __bf16* __restrict__ w16 = ws + OFF_WQKV;
  __bf16* __restrict__ qkv = ws + OFF_QKV;

  __shared__ __bf16 xt[TT * CCH];           // [t][c] bf16, 24KB
  for (int idx = threadIdx.x; idx < CCH * TT; idx += 256) {
    int c = idx >> 6, t = idx & 63;        // read x coalesced-ish over (c,t)
    xt[t * CCH + c] = (__bf16)x[(((size_t)n * CCH + c) * TT + t) * VV + vv];
  }
  __syncthreads();

  const int wave = threadIdx.x >> 5;
  const int l15  = threadIdx.x & 15;
  const int hi   = (threadIdx.x >> 4) & 1;

  for (int m = wave; m < 36; m += 8) {      // 36 m-tiles of 16 rows
    v8f acc[4] = {v8f{}, v8f{}, v8f{}, v8f{}};
    const int row = m * 16 + l15;
    #pragma unroll
    for (int kk = 0; kk < 6; ++kk) {
      const int kb = kk * 32;
      // A: two contiguous 8-half chunks (K = kb+hi*16.., kb+hi*16+8..)
      v8bf alo = *(const v8bf*)&w16[row * CCH + kb + hi * 16];
      v8bf ahi = *(const v8bf*)&w16[row * CCH + kb + hi * 16 + 8];
      v16bf a = cat8(alo, ahi);
      #pragma unroll
      for (int nt = 0; nt < 4; ++nt) {      // B: 16 contiguous halfs in LDS
        v16bf bf = *(const v16bf*)&xt[(nt * 16 + l15) * CCH + kb + hi * 16];
        acc[nt] = wmma_bf16(a, bf, acc[nt]);
      }
    }
    // store: o = m*16 + hi*8 + r (r=0..7 -> consecutive dd), one slot per tile
    const int obase = m * 16 + hi * 8;
    const int part  = (m * 16) / CCH;
    const int rem   = (m * 16) % CCH;
    const int slot  = part * 6 + (rem >> 5);
    const int ddb   = (rem & 31) + hi * 8;
    v8f bv = *(const v8f*)&bias[obase];
    #pragma unroll
    for (int nt = 0; nt < 4; ++nt) {
      const int t = nt * 16 + l15;
      if (part < 2) {                       // q,k: [slot][t][dd] contiguous
        v8bf vals;
        #pragma unroll
        for (int r = 0; r < 8; ++r) vals[r] = (__bf16)(acc[nt][r] + bv[r]);
        *(v8bf*)&qkv[(((size_t)b * 18 + slot) * TT + t) * DH + ddb] = vals;
      } else {                              // v: [slot][dd][t] transposed
        #pragma unroll
        for (int r = 0; r < 8; ++r)
          qkv[(((size_t)b * 18 + slot) * DH + ddb + r) * TT + t] =
              (__bf16)(acc[nt][r] + bv[r]);
      }
    }
  }
}

// ---------------- attention per (b,h) --------------------------------------
__global__ __launch_bounds__(128) void attn_kernel(__bf16* __restrict__ ws) {
  const int bh = blockIdx.x;
  const int b = bh / HEADS, h = bh % HEADS;
  const __bf16* __restrict__ qkv   = ws + OFF_QKV;
  const __bf16* __restrict__ rel16 = ws + OFF_REL;
  __bf16* __restrict__ attn        = ws + OFF_ATTN;

  const size_t base = (size_t)b * 18 * TT * DH;
  const __bf16* qp = qkv + base + (size_t)(h)      * TT * DH;  // [t][dd]
  const __bf16* kp = qkv + base + (size_t)(6 + h)  * TT * DH;  // [t][dd]
  const __bf16* vp = qkv + base + (size_t)(12 + h) * TT * DH;  // [dd][t]

  __shared__ float sbuf[4][16][128];        // per-wave staging, 32KB
  const int wave = threadIdx.x >> 5;
  const int l15  = threadIdx.x & 15;
  const int hi   = (threadIdx.x >> 4) & 1;
  const int tb   = wave * 16;
  float (*rb)[128] = sbuf[wave];

  // A fragment: q^T rows t=tb+l15; 16 contiguous halfs at hi*16
  v16bf aq = cat8(*(const v8bf*)&qp[(tb + l15) * DH + hi * 16],
                  *(const v8bf*)&qp[(tb + l15) * DH + hi * 16 + 8]);

  // logits = q^T k : 4 s-tiles, K=32 single step
  v8f lg[4];
  #pragma unroll
  for (int j = 0; j < 4; ++j) {
    v16bf kb = *(const v16bf*)&kp[(j * 16 + l15) * DH + hi * 16];
    lg[j] = wmma_bf16(aq, kb, v8f{});
  }

  // rel logits -> LDS (cols 0..127), rel_kT[dd][m] = rel16[m*32+dd]
  #pragma unroll
  for (int j = 0; j < 8; ++j) {
    v16bf rf = *(const v16bf*)&rel16[(j * 16 + l15) * DH + hi * 16];
    v8f rc = wmma_bf16(aq, rf, v8f{});
    #pragma unroll
    for (int r = 0; r < 8; ++r)
      rb[r + hi * 8][j * 16 + l15] = rc[r];
  }

  // rel_to_abs add: abs[t][s] = rel[t][s - t + 63]
  #pragma unroll
  for (int j = 0; j < 4; ++j)
    #pragma unroll
    for (int r = 0; r < 8; ++r) {
      int t = tb + r + hi * 8;
      int s = j * 16 + l15;
      lg[j][r] += rb[r + hi * 8][s - t + 63];
    }

  // softmax over s (row spread over 16 lanes x 4 tiles) — wave32 shuffles
  #pragma unroll
  for (int r = 0; r < 8; ++r) {
    float mx = fmaxf(fmaxf(lg[0][r], lg[1][r]), fmaxf(lg[2][r], lg[3][r]));
    #pragma unroll
    for (int msk = 1; msk < 16; msk <<= 1)
      mx = fmaxf(mx, __shfl_xor(mx, msk, 32));
    float e0 = __expf(lg[0][r] - mx), e1 = __expf(lg[1][r] - mx);
    float e2 = __expf(lg[2][r] - mx), e3 = __expf(lg[3][r] - mx);
    float sm = e0 + e1 + e2 + e3;
    #pragma unroll
    for (int msk = 1; msk < 16; msk <<= 1)
      sm += __shfl_xor(sm, msk, 32);
    float inv = __frcp_rn(sm);
    lg[0][r] = e0 * inv; lg[1][r] = e1 * inv;
    lg[2][r] = e2 * inv; lg[3][r] = e3 * inv;
  }

  // stage weights (C-layout) -> LDS, reload as A fragments (bf16)
  #pragma unroll
  for (int j = 0; j < 4; ++j)
    #pragma unroll
    for (int r = 0; r < 8; ++r)
      rb[r + hi * 8][j * 16 + l15] = lg[j][r];

  v8f oacc[2] = {v8f{}, v8f{}};
  #pragma unroll
  for (int kk = 0; kk < 2; ++kk) {
    // A reload: two contiguous 8-float LDS chunks, cvt to bf16
    v8f wlo = *(const v8f*)&rb[l15][kk * 32 + hi * 16];
    v8f whi = *(const v8f*)&rb[l15][kk * 32 + hi * 16 + 8];
    v16bf aw;
    #pragma unroll
    for (int i = 0; i < 8; ++i) {
      aw[i]     = (__bf16)wlo[i];
      aw[i + 8] = (__bf16)whi[i];
    }
    #pragma unroll
    for (int nt = 0; nt < 2; ++nt) {        // V is [dd][t]: contiguous in t
      v16bf vb = *(const v16bf*)&vp[(nt * 16 + l15) * TT + kk * 32 + hi * 16];
      oacc[nt] = wmma_bf16(aw, vb, oacc[nt]);
    }
  }

  // store with reference's reshape interleave folded into indices
  #pragma unroll
  for (int nt = 0; nt < 2; ++nt)
    #pragma unroll
    for (int r = 0; r < 8; ++r) {
      int t = tb + r + hi * 8;
      int dd = nt * 16 + l15;
      int flat2 = ((t & 1) << 10) + (dd << 5) + (t >> 1);
      int cc = flat2 >> 6, tout = flat2 & 63;
      attn[((size_t)b * TT + tout) * CCH + h * DH + cc] = (__bf16)oacc[nt][r];
    }
}

// ---------------- output projection: 192x192 x 192x64 per b ----------------
// LDS tile is filled by the Tensor Data Mover (1-D descriptor, 3072 x 8B),
// tracked with TENSORcnt instead of 96 VGPR round-trip loads.
__global__ __launch_bounds__(128) void out_gemm(
    const float* __restrict__ bias, const __bf16* __restrict__ ws,
    float* __restrict__ out) {
  const int b = blockIdx.x;
  const int n = b / VV, vv = b % VV;
  const __bf16* __restrict__ w16 = ws + OFF_WOUT;
  const __bf16* __restrict__ attn = ws + OFF_ATTN + (size_t)b * TT * CCH;

  __shared__ __bf16 at[TT * CCH];           // [t][c], 24KB, LDS offset 0

  if ((threadIdx.x >> 5) == 0) {            // one wave issues the TDM DMA
    unsigned long long ga = (unsigned long long)(const void*)attn;
    v4u g0;
    g0[0] = 1u;                                             // count=1
    g0[1] = 0u;                                             // lds_addr = 0
    g0[2] = (unsigned)(ga & 0xffffffffu);                   // global addr lo
    g0[3] = (unsigned)((ga >> 32) & 0x01ffffffu) | (2u << 30); // hi + type=2
    v8i g1;
    g1[0] = (int)(3u << 16);          // data_size = 8 bytes
    g1[1] = (int)(3072u << 16);       // tensor_dim0 = 3072 (lo16)
    g1[2] = (int)(1u << 16);          // tensor_dim0 hi = 0, tensor_dim1 = 1
    g1[3] = (int)(3072u << 16);       // tensor_dim1 hi = 0, tile_dim0 = 3072
    g1[4] = 1;                        // tile_dim1 = 1, tile_dim2 = 0
    g1[5] = 3072;                     // tensor_dim0_stride lo32
    g1[6] = 0;
    g1[7] = 0;
    v4i z = {0, 0, 0, 0};
#if __clang_major__ >= 23
    v8i z8 = {0, 0, 0, 0, 0, 0, 0, 0};
    __builtin_amdgcn_tensor_load_to_lds(g0, g1, z, z, z8, 0);
#else
    __builtin_amdgcn_tensor_load_to_lds(g0, g1, z, z, 0);
#endif
    __builtin_amdgcn_s_wait_tensorcnt(0);
  }
  __syncthreads();

  const int wave = threadIdx.x >> 5;
  const int l15  = threadIdx.x & 15;
  const int hi   = (threadIdx.x >> 4) & 1;

  for (int m = wave; m < 12; m += 4) {
    v8f acc[4] = {v8f{}, v8f{}, v8f{}, v8f{}};
    const int row = m * 16 + l15;
    #pragma unroll
    for (int kk = 0; kk < 6; ++kk) {
      const int kb = kk * 32;
      v16bf a = cat8(*(const v8bf*)&w16[row * CCH + kb + hi * 16],
                     *(const v8bf*)&w16[row * CCH + kb + hi * 16 + 8]);
      #pragma unroll
      for (int nt = 0; nt < 4; ++nt) {
        v16bf bf = *(const v16bf*)&at[(nt * 16 + l15) * CCH + kb + hi * 16];
        acc[nt] = wmma_bf16(a, bf, acc[nt]);
      }
    }
    const int obase = m * 16 + hi * 8;
    v8f bv = *(const v8f*)&bias[obase];
    #pragma unroll
    for (int nt = 0; nt < 4; ++nt) {
      const int t = nt * 16 + l15;
      #pragma unroll
      for (int r = 0; r < 8; ++r)
        out[(((size_t)n * CCH + obase + r) * TT + t) * VV + vv] =
            acc[nt][r] + bv[r];
    }
  }
}

extern "C" void kernel_launch(void* const* d_in, const int* in_sizes, int n_in,
                              void* d_out, int out_size, void* d_ws, size_t ws_size,
                              hipStream_t stream) {
  const float* x     = (const float*)d_in[0];
  const float* wqkv  = (const float*)d_in[1];
  const float* bqkv  = (const float*)d_in[2];
  const float* wout  = (const float*)d_in[3];
  const float* bout  = (const float*)d_in[4];
  const float* rel   = (const float*)d_in[5];
  float* out = (float*)d_out;
  __bf16* ws = (__bf16*)d_ws;

  prep_kernel<<<(O3 * CCH + 255) / 256, 256, 0, stream>>>(wqkv, wout, rel, ws);
  qkv_gemm<<<BB, 256, 0, stream>>>(x, bqkv, ws);
  attn_kernel<<<BB * HEADS, 128, 0, stream>>>(ws);
  out_gemm<<<BB, 128, 0, stream>>>(bout, ws, out);
}
